// CirculantLinear_11785390260446
// MI455X (gfx1250) — matrix operand order repaired and checked by reading, not compiled, not run on hardware
//
#include <hip/hip_runtime.h>

typedef __attribute__((ext_vector_type(16))) __bf16 v16bf;
typedef __attribute__((ext_vector_type(8)))  __bf16 v8bf;
typedef __attribute__((ext_vector_type(8)))  float  v8f;
typedef int v4i __attribute__((vector_size(16)));

// Problem dims (fixed by the reference)
constexpr int Bdim = 16384;   // batch
constexpr int Cdim = 2048;    // IN_C == OUT_C

// GEMM tiling
constexpr int BM = 128, BN = 128, BK = 64;
constexpr int LDA = BK + 8;   // 72 halfs -> 144B row stride (16B aligned, conflict-free)
constexpr int LDB = BK + 8;

#if defined(__gfx1250__) && __has_builtin(__builtin_amdgcn_global_load_async_to_lds_b128)
#define ASYNC_COPY 1
#else
#define ASYNC_COPY 0
#endif

__device__ __forceinline__ unsigned short f32_to_bf16_rne(float f) {
    unsigned int u = __float_as_uint(f);
    u += 0x7FFFu + ((u >> 16) & 1u);
    return (unsigned short)(u >> 16);
}

// ---- pack x: f32 -> bf16, 8 elements per thread ----------------------------
__global__ __launch_bounds__(256) void pack_x_bf16(const float* __restrict__ x,
                                                   unsigned short* __restrict__ xb) {
    size_t i = ((size_t)blockIdx.x * 256 + threadIdx.x) * 8;
    float4 a = *(const float4*)(x + i);
    float4 b = *(const float4*)(x + i + 4);
    uint4 o;
    o.x = (unsigned)f32_to_bf16_rne(a.x) | ((unsigned)f32_to_bf16_rne(a.y) << 16);
    o.y = (unsigned)f32_to_bf16_rne(a.z) | ((unsigned)f32_to_bf16_rne(a.w) << 16);
    o.z = (unsigned)f32_to_bf16_rne(b.x) | ((unsigned)f32_to_bf16_rne(b.y) << 16);
    o.w = (unsigned)f32_to_bf16_rne(b.z) | ((unsigned)f32_to_bf16_rne(b.w) << 16);
    *(uint4*)(xb + i) = o;
}

// ---- build W[n,k] = eigens[n/8, k/8, (n-k) mod 8], n-major bf16 ------------
__global__ __launch_bounds__(256) void pack_w_bf16(const float* __restrict__ eig,
                                                   unsigned short* __restrict__ wp) {
    int idx = blockIdx.x * 256 + threadIdx.x;       // 0 .. 2048*2048-1
    int n = idx >> 11;
    int k = idx & 2047;
    float e = eig[((size_t)((n >> 3) << 8) + (k >> 3)) * 8 + ((n - k) & 7)];
    wp[idx] = f32_to_bf16_rne(e);
}

#if ASYNC_COPY
__device__ __forceinline__ void async_copy16(void* lds, const void* g) {
    __builtin_amdgcn_global_load_async_to_lds_b128(
        (__attribute__((address_space(1))) v4i*)g,
        (__attribute__((address_space(3))) v4i*)lds,
        0, 0);
}
__device__ __forceinline__ void async_wait_all() {
#if __has_builtin(__builtin_amdgcn_s_wait_asynccnt)
    __builtin_amdgcn_s_wait_asynccnt(0);
#else
    asm volatile("s_wait_asynccnt 0x0" ::: "memory");
#endif
}
#endif

// 16 v_wmma_f32_16x16x32_bf16 per call (BK=64 -> two K-steps of 32)
__device__ __forceinline__ void compute_tile(
    const unsigned short* sa, const unsigned short* sb,
    int m_off, int n_off, int l16, int lhalf, v8f (&acc)[2][4])
{
    #pragma unroll
    for (int k0 = 0; k0 < BK; k0 += 32) {
        // A (16x32 MxK): lanes 0-15 hold K {0..7,16..23}, lanes 16-31 {8..15,24..31}
        v16bf afrag[2];
        #pragma unroll
        for (int mi = 0; mi < 2; ++mi) {
            const unsigned short* pa = &sa[(m_off + mi * 16 + l16) * LDA + k0 + lhalf * 8];
            union { v16bf v; v8bf h[2]; } ua;
            ua.h[0] = *(const v8bf*)pa;
            ua.h[1] = *(const v8bf*)(pa + 16);
            afrag[mi] = ua.v;
        }
        #pragma unroll
        for (int ni = 0; ni < 4; ++ni) {
            // B (32x16 KxN): lane = column, lanes 0-15 K {0..7,16..23}, lanes 16-31 {8..15,24..31}
            const unsigned short* pb = &sb[(n_off + ni * 16 + l16) * LDB + k0 + lhalf * 8];
            union { v16bf v; v8bf h[2]; } ub;
            ub.h[0] = *(const v8bf*)pb;
            ub.h[1] = *(const v8bf*)(pb + 16);
            v16bf bfrag = ub.v;
            #pragma unroll
            for (int mi = 0; mi < 2; ++mi) {
                acc[mi][ni] = __builtin_amdgcn_wmma_f32_16x16x32_bf16(
                    false, afrag[mi], false, bfrag,
                    (short)0, acc[mi][ni], false, false);
            }
        }
    }
}

// ---- main GEMM: y[M,N] = xb[M,K] * W[N,K]^T + bias, double-buffered --------
__global__ __launch_bounds__(256) void bcirc_gemm(
    const unsigned short* __restrict__ xb,   // [Bdim, Cdim] bf16, row-major
    const unsigned short* __restrict__ wp,   // [Cdim(N), Cdim(K)] bf16, n-major
    const float* __restrict__ bias,          // [Cdim]
    float* __restrict__ y)                   // [Bdim, Cdim] f32
{
    __shared__ unsigned short lds_a[2][BM * LDA];   // 2 x 18 KiB
    __shared__ unsigned short lds_b[2][BN * LDB];   // 2 x 18 KiB  (72 KiB of 320 KiB WGP LDS)

    const int bm = blockIdx.x & 127;         // 128 M-blocks (fast dim -> W tile L2 reuse)
    const int bn = blockIdx.x >> 7;          // 16 N-blocks
    const int m_base = bm * BM;
    const int n_base = bn * BN;

    const int tid   = threadIdx.x;
    const int wid   = tid >> 5;              // 8 waves
    const int lane  = tid & 31;
    const int m_off = (wid & 3) * 32;        // 4 waves along M
    const int n_off = (wid >> 2) * 64;       // 2 waves along N
    const int l16   = lane & 15;
    const int lhalf = (lane >> 4) & 1;

    // global->LDS: 16B chunk per thread, 8 chunks per 64-half row, 32 rows/pass
    const int c8 = (tid & 7) * 8;
    const int r0 = tid >> 3;
    const unsigned short* gA = xb + (size_t)(m_base + r0) * Cdim + c8;
    const unsigned short* gB = wp + (size_t)(n_base + r0) * Cdim + c8;
    const int la = r0 * LDA + c8;
    const int lb = r0 * LDB + c8;

    v8f acc[2][4];
    const v8f vzero = {0.f, 0.f, 0.f, 0.f, 0.f, 0.f, 0.f, 0.f};
    #pragma unroll
    for (int mi = 0; mi < 2; ++mi)
        #pragma unroll
        for (int ni = 0; ni < 4; ++ni)
            acc[mi][ni] = vzero;

#if ASYNC_COPY
    auto issue_tile = [&](int kt, int buf) {
        #pragma unroll
        for (int i = 0; i < 4; ++i) {
            async_copy16(&lds_a[buf][la + 32 * i * LDA], gA + (size_t)(32 * i) * Cdim + kt);
            async_copy16(&lds_b[buf][lb + 32 * i * LDB], gB + (size_t)(32 * i) * Cdim + kt);
        }
    };
    issue_tile(0, 0);
    async_wait_all();
    __syncthreads();
    int cur = 0;
    for (int kt = 0; kt < Cdim; kt += BK) {
        if (kt + BK < Cdim) issue_tile(kt + BK, cur ^ 1);   // prefetch next tile
        compute_tile(lds_a[cur], lds_b[cur], m_off, n_off, l16, lhalf, acc);
        async_wait_all();                                   // own async copies done
        __syncthreads();                                    // everyone's copies visible
        cur ^= 1;
    }
#else
    uint4 ra[4], rb[4];
    auto load_tile = [&](int kt) {
        #pragma unroll
        for (int i = 0; i < 4; ++i) {
            ra[i] = *(const uint4*)(gA + (size_t)(32 * i) * Cdim + kt);
            rb[i] = *(const uint4*)(gB + (size_t)(32 * i) * Cdim + kt);
        }
    };
    auto store_tile = [&](int buf) {
        #pragma unroll
        for (int i = 0; i < 4; ++i) {
            *(uint4*)(&lds_a[buf][la + 32 * i * LDA]) = ra[i];
            *(uint4*)(&lds_b[buf][lb + 32 * i * LDB]) = rb[i];
        }
    };
    load_tile(0);
    store_tile(0);
    __syncthreads();
    int cur = 0;
    for (int kt = 0; kt < Cdim; kt += BK) {
        bool more = (kt + BK) < Cdim;
        if (more) load_tile(kt + BK);     // overlap next tile's global loads with compute
        compute_tile(lds_a[cur], lds_b[cur], m_off, n_off, l16, lhalf, acc);
        if (more) store_tile(cur ^ 1);
        __syncthreads();
        cur ^= 1;
    }
#endif

    // Epilogue: C/D layout -> VGPR r holds M = r (lanes 0-15) / r+8 (lanes 16-31)
    #pragma unroll
    for (int ni = 0; ni < 4; ++ni) {
        int col = n_base + n_off + ni * 16 + l16;
        float bv = bias[col];
        #pragma unroll
        for (int mi = 0; mi < 2; ++mi) {
            #pragma unroll
            for (int r = 0; r < 8; ++r) {
                int row = m_base + m_off + mi * 16 + lhalf * 8 + r;
                y[(size_t)row * Cdim + col] = acc[mi][ni][r] + bv;
            }
        }
    }
}

extern "C" void kernel_launch(void* const* d_in, const int* in_sizes, int n_in,
                              void* d_out, int out_size, void* d_ws, size_t ws_size,
                              hipStream_t stream) {
    (void)in_sizes; (void)n_in; (void)out_size; (void)ws_size;
    const float* x    = (const float*)d_in[0];
    const float* eig  = (const float*)d_in[1];
    const float* bias = (const float*)d_in[2];
    float* y = (float*)d_out;

    unsigned short* xb = (unsigned short*)d_ws;                 // 16384*2048 bf16 = 64 MiB
    unsigned short* wp = xb + (size_t)Bdim * Cdim;              // 2048*2048 bf16 = 8 MiB

    pack_x_bf16<<<(Bdim * Cdim) / (256 * 8), 256, 0, stream>>>(x, xb);
    pack_w_bf16<<<(Cdim * Cdim) / 256, 256, 0, stream>>>(eig, wp);
    bcirc_gemm<<<(Bdim / BM) * (Cdim / BN), 256, 0, stream>>>(xb, wp, bias, y);
}
